// MultiHeadAttention_85899345981
// MI455X (gfx1250) — compile-verified, matrix operand order checked by reading, not run on hardware
//
#include <hip/hip_runtime.h>

// ---------------- problem constants ----------------
#define Bb 2
#define Ss 2048
#define Dd 1024
#define Hh 16
#define DEPTH 64          // Dd / Hh

#define DEV __device__ __forceinline__

typedef __attribute__((ext_vector_type(16))) _Float16 v16h;
typedef __attribute__((ext_vector_type(8)))  float    v8f;

// D = A(16x32 f16) * B(32x16 f16) + C(16x16 f32)   -> v_wmma_f32_16x16x32_f16
DEV v8f wmma16x16x32(v16h a, v16h b, v8f c) {
    return __builtin_amdgcn_wmma_f32_16x16x32_f16(
        /*neg_a=*/false, a, /*neg_b=*/false, b,
        /*c_mod=*/(short)0, c, /*reuse_a=*/false, /*reuse_b=*/false);
}

// Per-lane A/B fragment fill for 16x16x32 f16 WMMA.
// Lane l (l<16): row/col = l, K elements = {kb..kb+7, kb+16..kb+23} with kb = kk
// Lane l (l>=16): same row/col, kb = kk+8  (caller passes kb = kk + (lane>>4)*8)
DEV v16h frag_from_f16(const _Float16* __restrict__ p) {
    v16h r;
#pragma unroll
    for (int i = 0; i < 8; ++i) r[i] = p[i];
#pragma unroll
    for (int i = 0; i < 8; ++i) r[8 + i] = p[16 + i];
    return r;
}

DEV v16h frag_from_f32(const float* __restrict__ p) {
    v16h r;
#pragma unroll
    for (int i = 0; i < 8; ++i) r[i] = (_Float16)p[i];
#pragma unroll
    for (int i = 0; i < 8; ++i) r[8 + i] = (_Float16)p[16 + i];
    return r;
}

// ---------------- kernel 1: weight convert + transpose (fp32 [r][c] -> f16 [c][r]) ----
__global__ void mha_convert_weights(const float* __restrict__ wq, const float* __restrict__ wk,
                                    const float* __restrict__ wv, const float* __restrict__ wo,
                                    _Float16* __restrict__ wqT, _Float16* __restrict__ wkT,
                                    _Float16* __restrict__ wvT, _Float16* __restrict__ woT)
{
    int z = blockIdx.z;
    const float* src = (z == 0) ? wq : (z == 1) ? wk : (z == 2) ? wv : wo;
    _Float16*    dst = (z == 0) ? wqT : (z == 1) ? wkT : (z == 2) ? wvT : woT;
    int idx = blockIdx.x * 256 + threadIdx.x;      // coalesced read over fast col index
    int r = idx >> 10;
    int c = idx & (Dd - 1);
    dst[(size_t)c * Dd + r] = (_Float16)src[idx];
}

// ---------------- kernel 2: QKV projection (one 16x64 strip / wave) ----------
// z=0: Qh = q@wq+bq  -> qh  [bh][s][d]   f16
// z=1: Kh = k@wk+bk  -> kh  [bh][s][d]   f16
// z=2: Vh = v@wv+bv  -> vhT [bh][d][s]   f16 (transposed for ctx B-fragments)
// Column group g covers cols [g*64, g*64+64) == head g exactly.
__global__ void mha_qkv_proj(const float* __restrict__ q, const float* __restrict__ k,
                             const float* __restrict__ v,
                             const _Float16* __restrict__ wqT, const _Float16* __restrict__ wkT,
                             const _Float16* __restrict__ wvT,
                             const float* __restrict__ bq, const float* __restrict__ bk,
                             const float* __restrict__ bv,
                             _Float16* __restrict__ qh, _Float16* __restrict__ kh,
                             _Float16* __restrict__ vhT)
{
    int z = blockIdx.z;
    const float*    X  = (z == 0) ? q   : (z == 1) ? k   : v;
    const _Float16* WT = (z == 0) ? wqT : (z == 1) ? wkT : wvT;
    const float*    bi = (z == 0) ? bq  : (z == 1) ? bk  : bv;

    int lane = threadIdx.x & 31;
    int wid  = threadIdx.x >> 5;
    int t  = blockIdx.x * 8 + wid;            // 4096 waves per z
    int tm = t >> 4;                          // row tile: 0..255  (B*S/16)
    int g  = t & 15;                          // 64-col group == head id
    int half = lane >> 4;
    int l15  = lane & 15;

    int arow = tm * 16 + l15;                 // flat row b*S + s
    const float*    ap = X  + (size_t)arow * Dd;
    const _Float16* bp = WT + (size_t)(g * 64 + l15) * Dd;

    v8f acc0 = {}, acc1 = {}, acc2 = {}, acc3 = {};
    for (int kk = 0; kk < Dd; kk += 32) {
        int kb = kk + half * 8;
        __builtin_prefetch(ap + kb + 64, 0, 0);
        v16h a  = frag_from_f32(ap + kb);
        v16h b0 = frag_from_f16(bp + (size_t)(0 * 16) * Dd + kb);
        acc0 = wmma16x16x32(a, b0, acc0);
        v16h b1 = frag_from_f16(bp + (size_t)(1 * 16) * Dd + kb);
        acc1 = wmma16x16x32(a, b1, acc1);
        v16h b2 = frag_from_f16(bp + (size_t)(2 * 16) * Dd + kb);
        acc2 = wmma16x16x32(a, b2, acc2);
        v16h b3 = frag_from_f16(bp + (size_t)(3 * 16) * Dd + kb);
        acc3 = wmma16x16x32(a, b3, acc3);
    }

    float bb0 = bi[g * 64 + 0  + l15];
    float bb1 = bi[g * 64 + 16 + l15];
    float bb2 = bi[g * 64 + 32 + l15];
    float bb3 = bi[g * 64 + 48 + l15];
#pragma unroll
    for (int j = 0; j < 8; ++j) {
        int r    = tm * 16 + half * 8 + j;    // flat b*S + s
        int bidx = r >> 11;                   // / S
        int s    = r & (Ss - 1);
        float v0 = acc0[j] + bb0, v1 = acc1[j] + bb1;
        float v2 = acc2[j] + bb2, v3 = acc3[j] + bb3;
        if (z == 2) {
            _Float16* dstp = vhT + ((size_t)(bidx * Hh + g) * DEPTH + l15) * Ss + s;
            dstp[(size_t)(0)      * Ss] = (_Float16)v0;
            dstp[(size_t)(16)     * Ss] = (_Float16)v1;
            dstp[(size_t)(32)     * Ss] = (_Float16)v2;
            dstp[(size_t)(48)     * Ss] = (_Float16)v3;
        } else {
            _Float16* dst = ((z == 0) ? qh : kh)
                          + ((size_t)(bidx * Hh + g) * Ss + s) * DEPTH + l15;
            dst[0]  = (_Float16)v0;
            dst[16] = (_Float16)v1;
            dst[32] = (_Float16)v2;
            dst[48] = (_Float16)v3;
        }
    }
}

// ---------------- kernel 3: fused logits + mask + softmax --------------------
// One block owns a full 16-query strip: logits kept in a 128 KB LDS buffer
// (CDNA5: up to 320 KB LDS per workgroup), softmaxed in place, written once.
__global__ __launch_bounds__(256) void mha_scores_softmax(
    const _Float16* __restrict__ qh, const _Float16* __restrict__ kh,
    const float* __restrict__ mask, float* __restrict__ attn)
{
    __shared__ float lds[16 * Ss];            // 128 KB logit strip
    __shared__ float red[256];
    __shared__ float rowinv[16];

    int tid  = threadIdx.x;
    int lane = tid & 31;
    int wid  = tid >> 5;
    int blk  = blockIdx.x;                    // 0..4095
    int bh   = blk >> 7;                      // / (S/16)
    int qt   = blk & 127;
    int half = lane >> 4;
    int l15  = lane & 15;

    // A fragments (query rows) loaded once, reused for all 16 k-tiles
    const _Float16* qb = qh + (size_t)bh * Ss * DEPTH + (size_t)(qt * 16 + l15) * DEPTH;
    v16h a0 = frag_from_f16(qb + half * 8);
    v16h a1 = frag_from_f16(qb + 32 + half * 8);

    int bidx = bh >> 4;                       // / H
    const float* mrow = mask + (size_t)bidx * Ss * Ss;
    const _Float16* khb = kh + (size_t)bh * Ss * DEPTH;

    for (int kt = wid; kt < Ss / 16; kt += 8) {
        const _Float16* kbp = khb + (size_t)(kt * 16 + l15) * DEPTH;
        v16h b0 = frag_from_f16(kbp + half * 8);
        v16h b1 = frag_from_f16(kbp + 32 + half * 8);
        v8f acc = {};
        acc = wmma16x16x32(a0, b0, acc);
        acc = wmma16x16x32(a1, b1, acc);
        int key = kt * 16 + l15;
#pragma unroll
        for (int j = 0; j < 8; ++j) {
            int m  = half * 8 + j;            // local query row 0..15
            int qq = qt * 16 + m;
            float mval = mrow[(size_t)qq * Ss + key];
            lds[m * Ss + key] = acc[j] * 0.125f - 1e9f * mval;   // 1/sqrt(64)
        }
    }
    __syncthreads();

    // row softmax: 16 threads per row
    int row = tid >> 4;
    int sub = tid & 15;
    float mx = -3.0e38f;
    for (int i = sub; i < Ss; i += 16) mx = fmaxf(mx, lds[row * Ss + i]);
    red[tid] = mx;
    __syncthreads();
#pragma unroll
    for (int ofs = 8; ofs > 0; ofs >>= 1) {
        if (sub < ofs) red[tid] = fmaxf(red[tid], red[tid + ofs]);
        __syncthreads();
    }
    mx = red[row * 16];
    __syncthreads();

    float sum = 0.0f;
    for (int i = sub; i < Ss; i += 16) {
        float e = __expf(lds[row * Ss + i] - mx);
        lds[row * Ss + i] = e;
        sum += e;
    }
    red[tid] = sum;
    __syncthreads();
#pragma unroll
    for (int ofs = 8; ofs > 0; ofs >>= 1) {
        if (sub < ofs) red[tid] += red[tid + ofs];
        __syncthreads();
    }
    if (sub == 0) rowinv[row] = 1.0f / red[row * 16];
    __syncthreads();

    // coalesced single write of the normalized strip
    float* orow = attn + (size_t)bh * Ss * Ss + (size_t)(qt * 16) * Ss;
    for (int i = tid; i < 16 * Ss; i += 256) {
        orow[i] = lds[i] * rowinv[i >> 11];
    }
}

// ---------------- kernel 4: ctx = attn @ Vh (one 16x64 strip / wave) ---------
__global__ void mha_ctx(const float* __restrict__ attn, const _Float16* __restrict__ vhT,
                        _Float16* __restrict__ ctx)
{
    int lane = threadIdx.x & 31;
    int wid  = threadIdx.x >> 5;
    int t  = blockIdx.x * 8 + wid;            // 4096 waves total
    int bh = t >> 7;                          // / 128
    int qt = t & 127;
    int half = lane >> 4;
    int l15  = lane & 15;

    const float*    ap = attn + (size_t)bh * Ss * Ss + (size_t)(qt * 16 + l15) * Ss;
    const _Float16* vb = vhT  + (size_t)bh * DEPTH * Ss + (size_t)l15 * Ss;

    v8f acc0 = {}, acc1 = {}, acc2 = {}, acc3 = {};
    for (int kk = 0; kk < Ss; kk += 32) {
        int kb = kk + half * 8;
        __builtin_prefetch(ap + kb + 64, 0, 0);
        v16h a  = frag_from_f32(ap + kb);     // attn probs -> f16, loaded once per strip
        v16h b0 = frag_from_f16(vb + (size_t)(0)  * Ss + kb);
        acc0 = wmma16x16x32(a, b0, acc0);
        v16h b1 = frag_from_f16(vb + (size_t)(16) * Ss + kb);
        acc1 = wmma16x16x32(a, b1, acc1);
        v16h b2 = frag_from_f16(vb + (size_t)(32) * Ss + kb);
        acc2 = wmma16x16x32(a, b2, acc2);
        v16h b3 = frag_from_f16(vb + (size_t)(48) * Ss + kb);
        acc3 = wmma16x16x32(a, b3, acc3);
    }

    int bidx = bh >> 4;
    int h    = bh & (Hh - 1);
#pragma unroll
    for (int j = 0; j < 8; ++j) {
        int qq = qt * 16 + half * 8 + j;
        _Float16* cp = ctx + (size_t)(bidx * Ss + qq) * Dd + h * DEPTH + l15;
        cp[0]  = (_Float16)acc0[j];
        cp[16] = (_Float16)acc1[j];
        cp[32] = (_Float16)acc2[j];
        cp[48] = (_Float16)acc3[j];
    }
}

// ---------------- kernel 5: out = ctx @ wo + bo (one 16x64 strip / wave) -----
__global__ void mha_out_proj(const _Float16* __restrict__ ctx, const _Float16* __restrict__ woT,
                             const float* __restrict__ bo, float* __restrict__ out)
{
    int lane = threadIdx.x & 31;
    int wid  = threadIdx.x >> 5;
    int t  = blockIdx.x * 8 + wid;            // 4096 waves
    int tm = t >> 4;
    int g  = t & 15;
    int half = lane >> 4;
    int l15  = lane & 15;

    int arow = tm * 16 + l15;
    const _Float16* ap = ctx + (size_t)arow * Dd;
    const _Float16* bp = woT + (size_t)(g * 64 + l15) * Dd;

    v8f acc0 = {}, acc1 = {}, acc2 = {}, acc3 = {};
    for (int kk = 0; kk < Dd; kk += 32) {
        int kb = kk + half * 8;
        v16h a  = frag_from_f16(ap + kb);
        v16h b0 = frag_from_f16(bp + (size_t)(0 * 16) * Dd + kb);
        acc0 = wmma16x16x32(a, b0, acc0);
        v16h b1 = frag_from_f16(bp + (size_t)(1 * 16) * Dd + kb);
        acc1 = wmma16x16x32(a, b1, acc1);
        v16h b2 = frag_from_f16(bp + (size_t)(2 * 16) * Dd + kb);
        acc2 = wmma16x16x32(a, b2, acc2);
        v16h b3 = frag_from_f16(bp + (size_t)(3 * 16) * Dd + kb);
        acc3 = wmma16x16x32(a, b3, acc3);
    }

    float bb0 = bo[g * 64 + 0  + l15];
    float bb1 = bo[g * 64 + 16 + l15];
    float bb2 = bo[g * 64 + 32 + l15];
    float bb3 = bo[g * 64 + 48 + l15];
#pragma unroll
    for (int j = 0; j < 8; ++j) {
        int r = tm * 16 + half * 8 + j;
        float* op = out + (size_t)r * Dd + g * 64 + l15;
        op[0]  = acc0[j] + bb0;
        op[16] = acc1[j] + bb1;
        op[32] = acc2[j] + bb2;
        op[48] = acc3[j] + bb3;
    }
}

// ---------------- host launch ------------------------------------------------
extern "C" void kernel_launch(void* const* d_in, const int* in_sizes, int n_in,
                              void* d_out, int out_size, void* d_ws, size_t ws_size,
                              hipStream_t stream)
{
    (void)in_sizes; (void)n_in; (void)out_size; (void)ws_size;

    const float* v    = (const float*)d_in[0];
    const float* k    = (const float*)d_in[1];
    const float* q    = (const float*)d_in[2];
    const float* mask = (const float*)d_in[3];
    const float* wq   = (const float*)d_in[4];
    const float* bq   = (const float*)d_in[5];
    const float* wk   = (const float*)d_in[6];
    const float* bk   = (const float*)d_in[7];
    const float* wv   = (const float*)d_in[8];
    const float* bv   = (const float*)d_in[9];
    const float* wo   = (const float*)d_in[10];
    const float* bo   = (const float*)d_in[11];

    float* out  = (float*)d_out;
    float* attn = out + (size_t)Bb * Ss * Dd;       // tuple output #2

    // workspace carve-up (f16): 4 transposed weights + qh + kh + vhT + ctx  (~40 MB)
    _Float16* ws  = (_Float16*)d_ws;
    _Float16* wqT = ws;
    _Float16* wkT = wqT + (size_t)Dd * Dd;
    _Float16* wvT = wkT + (size_t)Dd * Dd;
    _Float16* woT = wvT + (size_t)Dd * Dd;
    _Float16* qh  = woT + (size_t)Dd * Dd;
    _Float16* kh  = qh  + (size_t)Bb * Hh * Ss * DEPTH;
    _Float16* vhT = kh  + (size_t)Bb * Hh * Ss * DEPTH;
    _Float16* ctx = vhT + (size_t)Bb * Hh * Ss * DEPTH;

    // 1. weights fp32 -> f16, transposed
    mha_convert_weights<<<dim3((Dd * Dd) / 256, 1, 4), 256, 0, stream>>>(
        wq, wk, wv, wo, wqT, wkT, wvT, woT);

    // 2. QKV projections (z selects q/k/v); 4096 waves per z
    mha_qkv_proj<<<dim3(512, 1, 3), 256, 0, stream>>>(
        q, k, v, wqT, wkT, wvT, bq, bk, bv, qh, kh, vhT);

    // 3. fused logits + mask + softmax, one block per 16-query strip
    mha_scores_softmax<<<dim3(Bb * Hh * (Ss / 16)), 256, 0, stream>>>(qh, kh, mask, attn);

    // 4. ctx = attn @ Vh; 4096 waves
    mha_ctx<<<dim3(512), 256, 0, stream>>>(attn, vhT, ctx);

    // 5. output projection; 4096 waves
    mha_out_proj<<<dim3(512), 256, 0, stream>>>(ctx, woT, bo, out);
}
